// PainnModel_variational_59141699666427
// MI455X (gfx1250) — matrix operand order repaired and checked by reading, not compiled, not run on hardware
//
#include <hip/hip_runtime.h>
#include <hip/hip_bf16.h>
#include <math.h>

// ---------------- problem constants ----------------
#define Bc    8
#define Nc    2000
#define Ec    30000
#define Hc    128
#define Lc    3
#define EMBc  20
#define NTOT  (Bc * Nc)        // 16000
#define ETOT  (Bc * Ec)        // 240000
#define CUT   5.0f
#define PI_F  3.14159265358979f

typedef __attribute__((ext_vector_type(16))) __bf16 bf16x16;
typedef __attribute__((ext_vector_type(8)))  __bf16 bf16x8;
typedef __attribute__((ext_vector_type(4)))  __bf16 bf16x4;
typedef __attribute__((ext_vector_type(8)))  float  f32x8;

// ---------------- WMMA fragment helpers (wave32, 16x16x32 bf16) ---------------
__device__ __forceinline__ bf16x16 frag_from(const __bf16* p, int hiOff) {
  bf16x8 lo = *(const bf16x8*)p;
  bf16x8 hi = *(const bf16x8*)(p + hiOff);
  return __builtin_shufflevector(lo, hi, 0, 1, 2, 3, 4, 5, 6, 7,
                                 8, 9, 10, 11, 12, 13, 14, 15);
}

// A: 16xK row-major tile. Lane m = lane&15; lane<16 -> K {0..7,16..23},
// lane>=16 -> K {8..15,24..31}.  Per-lane base precomputed by caller.
__device__ __forceinline__ const __bf16* a_lane_base(const __bf16* A, int lda) {
  int lane = threadIdx.x & 31;
  return A + (lane & 15) * lda + ((lane & 16) ? 8 : 0);
}

// B packed: per (kt,nt) tile, lane's 16 bf16 contiguous at ((kt*ntiles+nt)*32+lane)*16.
__device__ __forceinline__ const __bf16* b_lane_base(const __bf16* P, int nt0) {
  int lane = threadIdx.x & 31;
  return P + (((size_t)nt0 * 32 + lane) << 4);
}

__device__ __forceinline__ f32x8 wmma_bf16(bf16x16 a, bf16x16 b, f32x8 c) {
  return __builtin_amdgcn_wmma_f32_16x16x32_bf16(false, a, false, b, (short)0, c,
                                                 false, false);
}

__device__ __forceinline__ float silu(float x) {
  return x * __builtin_amdgcn_rcpf(1.0f + __expf(-x));
}

// ---------------- pack f32 row-major KxN weight into bf16 WMMA fragments ------
__global__ void pack_b_kernel(const float* __restrict__ src, __bf16* __restrict__ out,
                              int Kp, int N, int Ksrc) {
  int idx = blockIdx.x * 256 + threadIdx.x;
  int total = Kp * N;
  if (idx >= total) return;
  int e    = idx & 15;
  int lane = (idx >> 4) & 31;
  int tile = idx >> 9;
  int ntiles = N >> 4;
  int kt = tile / ntiles;
  int nt = tile - kt * ntiles;
  int n = nt * 16 + (lane & 15);
  int k = kt * 32 + ((lane & 16) ? 8 : 0) + (e & 7) + ((e & 8) ? 16 : 0);
  float v = (k < Ksrc) ? src[k * N + n] : 0.f;
  out[idx] = (__bf16)v;
}

// ---------------- generic WMMA GEMM: C = act(A @ B + bias) --------------------
// One wave computes a 16x128 strip (8 tiles); grid.y selects the N-strip.
__global__ void gemm_wmma_kernel(const __bf16* __restrict__ A,
                                 const __bf16* __restrict__ Bp,
                                 const float* __restrict__ bias,
                                 float* __restrict__ outF,
                                 __bf16* __restrict__ outB,
                                 int N, int K, int applySilu, int Mtiles) {
  int wave = threadIdx.x >> 5;
  int lane = threadIdx.x & 31;
  int mt = blockIdx.x * (blockDim.x >> 5) + wave;
  if (mt >= Mtiles) return;
  int ns = blockIdx.y;
  int ntiles = N >> 4;

  const __bf16* ap = a_lane_base(A + (long long)mt * 16 * K, K);
  const __bf16* bp = b_lane_base(Bp, ns * 8);
  size_t kstride = (size_t)ntiles * 512;  // elements per kt step

  f32x8 acc[8];
#pragma unroll
  for (int j = 0; j < 8; ++j) acc[j] = (f32x8){0.f, 0.f, 0.f, 0.f, 0.f, 0.f, 0.f, 0.f};

  for (int k0 = 0; k0 < K; k0 += 32) {
    bf16x16 af = frag_from(ap, 16);
    ap += 32;
#pragma unroll
    for (int j = 0; j < 8; ++j) {
      acc[j] = wmma_bf16(af, frag_from(bp + j * 512, 8), acc[j]);
    }
    bp += kstride;
  }

  int mbase = (lane & 16) ? 8 : 0;
#pragma unroll
  for (int j = 0; j < 8; ++j) {
    int n = (ns * 8 + j) * 16 + (lane & 15);
    float bv = bias ? bias[n] : 0.f;
#pragma unroll
    for (int r = 0; r < 8; ++r) {
      long long row = (long long)mt * 16 + mbase + r;
      float x = acc[j][r] + bv;
      if (applySilu) x = silu(x);
      long long o = row * N + n;
      if (outF) outF[o] = x;
      if (outB) outB[o] = (__bf16)x;
    }
  }
}

// ---------------- edge geometry + sin basis (padded K=32, bf16) ----------------
__global__ void edge_geom_kernel(const float* __restrict__ xyz,
                                 const int* __restrict__ edges,
                                 const float* __restrict__ edisp,
                                 const float* __restrict__ cell,
                                 float* __restrict__ unit,
                                 float* __restrict__ fc,
                                 __bf16* __restrict__ esB) {
  int eg = blockIdx.x * 256 + threadIdx.x;
  if (eg >= ETOT) return;
  int b = eg / Ec;
  int sn = edges[eg * 2 + 0] + b * Nc;
  int dn = edges[eg * 2 + 1] + b * Nc;
  float diff[3];
  float d2 = 0.f;
#pragma unroll
  for (int k = 0; k < 3; ++k) {
    float disp = 0.f;
#pragma unroll
    for (int j = 0; j < 3; ++j) disp += edisp[eg * 3 + j] * cell[(b * 3 + j) * 3 + k];
    float df = xyz[dn * 3 + k] + disp - xyz[sn * 3 + k];
    diff[k] = df;
    d2 += df * df;
  }
  float dist = sqrtf(d2);
  float inv = 1.0f / dist;
#pragma unroll
  for (int k = 0; k < 3; ++k) unit[eg * 3 + k] = diff[k] * inv;
  fc[eg] = 0.5f * (cosf(PI_F * dist / CUT) + 1.0f);
  float w = PI_F * dist / CUT;
  bf16x8 es[4];
#pragma unroll
  for (int j = 0; j < 32; ++j) {
    float v = (j < EMBc) ? (sinf((float)(j + 1) * w) * inv) : 0.f;
    es[j >> 3][j & 7] = (__bf16)v;
  }
  bf16x8* eo = (bf16x8*)(esB + (long long)eg * 32);
#pragma unroll
  for (int j = 0; j < 4; ++j) eo[j] = es[j];
}

// ---------------- small utility kernels (wide) ----------------
__global__ void zero_f32_kernel(float* p, long long n4) {  // n4 = n/4
  long long i = (long long)blockIdx.x * 256 + threadIdx.x;
  if (i < n4) ((float4*)p)[i] = make_float4(0.f, 0.f, 0.f, 0.f);
}

__device__ __forceinline__ bf16x8 cvt8(float4 a, float4 b) {
  bf16x8 r;
  r[0] = (__bf16)a.x; r[1] = (__bf16)a.y; r[2] = (__bf16)a.z; r[3] = (__bf16)a.w;
  r[4] = (__bf16)b.x; r[5] = (__bf16)b.y; r[6] = (__bf16)b.z; r[7] = (__bf16)b.w;
  return r;
}

__global__ void cast_f32_bf16_kernel(const float* __restrict__ src,
                                     __bf16* __restrict__ dst, long long n8) {  // n/8
  long long i = (long long)blockIdx.x * 256 + threadIdx.x;
  if (i >= n8) return;
  const float4* s4 = (const float4*)src;
  ((bf16x8*)dst)[i] = cvt8(s4[i * 2], s4[i * 2 + 1]);
}

__global__ void embed_kernel(const int* __restrict__ nodes,
                             const float* __restrict__ emb,
                             float* __restrict__ s) {
  int i = blockIdx.x * 256 + threadIdx.x;  // NTOT*32 (4 ch each)
  if (i >= NTOT * 32) return;
  int n = i >> 5, c = i & 31;
  ((float4*)s)[n * 32 + c] = ((const float4*)emb)[nodes[n] * 32 + c];
}

__global__ void gather_cast_kernel(const float* __restrict__ s,
                                   const int* __restrict__ edges,
                                   __bf16* __restrict__ sgB) {
  long long i = (long long)blockIdx.x * 256 + threadIdx.x;  // ETOT*16 (8 ch each)
  if (i >= (long long)ETOT * 16) return;
  int eg = (int)(i >> 4);
  int c  = (int)(i & 15);
  int b  = eg / Ec;
  int sn = edges[eg * 2 + 0] + b * Nc;
  const float4* s4 = (const float4*)(s + sn * Hc);
  ((bf16x8*)sgB)[i] = cvt8(s4[c * 2], s4[c * 2 + 1]);
}

// ---------------- fused edge kernel -------------------------------------------
// m = (t@m2w + b2) * ((es@fw + fb)*fc); split g_sv/g_ev/g_s; mv from v_old[src];
// scatter-add to s,v at dst. Block = 16 edges, 8 waves, m tile in LDS only.
__global__ void edge_message_kernel(const __bf16* __restrict__ tB,   // ETOT x 128
                                    const __bf16* __restrict__ esB,  // ETOT x 32
                                    const __bf16* __restrict__ m2p,  // packed 128x384
                                    const float* __restrict__ m2b,   // 384
                                    const __bf16* __restrict__ fwp,  // packed 32x384
                                    const float* __restrict__ fb,    // 384
                                    const float* __restrict__ fc,    // ETOT
                                    const float* __restrict__ unit,  // ETOT x 3
                                    const float* __restrict__ v_old, // NTOT x 3 x 128
                                    const int* __restrict__ edges,
                                    float* __restrict__ s_acc,       // NTOT x 128
                                    float* __restrict__ v_acc) {     // NTOT x 3 x 128
  __shared__ float mt[16 * 384];
  __shared__ float su[16 * 3];
  __shared__ int   ssn[16], sdn[16];
  int e0   = blockIdx.x * 16;
  int wave = threadIdx.x >> 5;
  int lane = threadIdx.x & 31;

  if (threadIdx.x < 48) su[threadIdx.x] = unit[e0 * 3 + threadIdx.x];
  if (threadIdx.x < 16) {
    int eg = e0 + threadIdx.x;
    int b = eg / Ec;
    ssn[threadIdx.x] = edges[eg * 2 + 0] + b * Nc;
    sdn[threadIdx.x] = edges[eg * 2 + 1] + b * Nc;
  }

  const __bf16* ap = a_lane_base(tB + (long long)e0 * 128, 128);
  const __bf16* bp = b_lane_base(m2p, wave * 3);
  const size_t kstride = 24 * 512;

  // main GEMM: 3 tiles per wave, A fragment reused across tiles
  f32x8 acc[3];
#pragma unroll
  for (int t = 0; t < 3; ++t) acc[t] = (f32x8){0.f, 0.f, 0.f, 0.f, 0.f, 0.f, 0.f, 0.f};
#pragma unroll
  for (int kt = 0; kt < 4; ++kt) {
    bf16x16 af = frag_from(ap, 16);
    ap += 32;
#pragma unroll
    for (int t = 0; t < 3; ++t) {
      acc[t] = wmma_bf16(af, frag_from(bp + t * 512, 8), acc[t]);
    }
    bp += kstride;
  }
  // filter GEMM (single K step) + epilogue
  bf16x16 ef = frag_from(a_lane_base(esB + (long long)e0 * 32, 32), 16);
  const __bf16* fp = b_lane_base(fwp, wave * 3);
  int mbase = (lane & 16) ? 8 : 0;
#pragma unroll
  for (int t = 0; t < 3; ++t) {
    f32x8 facc = (f32x8){0.f, 0.f, 0.f, 0.f, 0.f, 0.f, 0.f, 0.f};
    facc = wmma_bf16(ef, frag_from(fp + t * 512, 8), facc);
    int n = (wave * 3 + t) * 16 + (lane & 15);
    float b2 = m2b[n], fbv = fb[n];
#pragma unroll
    for (int r = 0; r < 8; ++r) {
      int m = mbase + r;
      mt[m * 384 + n] = (acc[t][r] + b2) * ((facc[r] + fbv) * fc[e0 + m]);
    }
  }
  __syncthreads();

  // scatter: 16 edges x 128 channels
  for (int idx = threadIdx.x; idx < 16 * 128; idx += 256) {
    int i = idx >> 7, h = idx & 127;
    int sn = ssn[i], dn = sdn[i];
    float g_sv = mt[i * 384 + h];
    float g_ev = mt[i * 384 + 128 + h];
    float g_s  = mt[i * 384 + 256 + h];
    atomicAdd(&s_acc[dn * Hc + h], g_s);
#pragma unroll
    for (int d = 0; d < 3; ++d) {
      float mvv = v_old[(sn * 3 + d) * Hc + h] * g_sv + g_ev * su[i * 3 + d];
      atomicAdd(&v_acc[(dn * 3 + d) * Hc + h], mvv);
    }
  }
}

// ---------------- node-stage pre: Vn, cat=[s,Vn] (bf16), U.V (4-wide) --------
__global__ void node_pre_kernel(const float* __restrict__ s,
                                const float* __restrict__ U,
                                const float* __restrict__ V,
                                float* __restrict__ UdotV,
                                __bf16* __restrict__ catB) {
  int i = blockIdx.x * 256 + threadIdx.x;  // NTOT*32 (4 ch each)
  if (i >= NTOT * 32) return;
  int n = i >> 5, c = i & 31;
  float4 vn2 = make_float4(1e-12f, 1e-12f, 1e-12f, 1e-12f);
  float4 dot = make_float4(0.f, 0.f, 0.f, 0.f);
#pragma unroll
  for (int d = 0; d < 3; ++d) {
    float4 uv = ((const float4*)U)[(n * 3 + d) * 32 + c];
    float4 vv = ((const float4*)V)[(n * 3 + d) * 32 + c];
    vn2.x += vv.x * vv.x; vn2.y += vv.y * vv.y;
    vn2.z += vv.z * vv.z; vn2.w += vv.w * vv.w;
    dot.x += uv.x * vv.x; dot.y += uv.y * vv.y;
    dot.z += uv.z * vv.z; dot.w += uv.w * vv.w;
  }
  ((float4*)UdotV)[i] = dot;
  float4 sv = ((const float4*)s)[i];
  bf16x4 cs, cn;
  cs[0] = (__bf16)sv.x; cs[1] = (__bf16)sv.y; cs[2] = (__bf16)sv.z; cs[3] = (__bf16)sv.w;
  cn[0] = (__bf16)sqrtf(vn2.x); cn[1] = (__bf16)sqrtf(vn2.y);
  cn[2] = (__bf16)sqrtf(vn2.z); cn[3] = (__bf16)sqrtf(vn2.w);
  ((bf16x4*)(catB + n * 256))[c]      = cs;
  ((bf16x4*)(catB + n * 256 + 128))[c] = cn;
}

// ---------------- node-stage update (4-wide) ----------------
__global__ void node_update_kernel(const float* __restrict__ a,     // NTOT x 384
                                   const float* __restrict__ U,
                                   const float* __restrict__ UdotV,
                                   float* __restrict__ s,
                                   float* __restrict__ v) {
  int i = blockIdx.x * 256 + threadIdx.x;  // NTOT*32 (4 ch each)
  if (i >= NTOT * 32) return;
  int n = i >> 5, c = i & 31;
  const float4* a4 = (const float4*)(a + n * 384);
  float4 avv = a4[c];
  float4 asv = a4[32 + c];
  float4 ass = a4[64 + c];
  float4 udv = ((const float4*)UdotV)[i];
  float4 sv = ((float4*)s)[i];
  sv.x += ass.x + asv.x * udv.x; sv.y += ass.y + asv.y * udv.y;
  sv.z += ass.z + asv.z * udv.z; sv.w += ass.w + asv.w * udv.w;
  ((float4*)s)[i] = sv;
#pragma unroll
  for (int d = 0; d < 3; ++d) {
    int o = (n * 3 + d) * 32 + c;
    float4 uv = ((const float4*)U)[o];
    float4 vv = ((float4*)v)[o];
    vv.x += avv.x * uv.x; vv.y += avv.y * uv.y;
    vv.z += avv.z * uv.z; vv.w += avv.w * uv.w;
    ((float4*)v)[o] = vv;
  }
}

// ---------------- readout: per-row dot with r2_w, batch sum ----------------
__global__ void readout_kernel(const __bf16* __restrict__ t2,  // NTOT x 128
                               const float* __restrict__ r2w,  // 128
                               const float* __restrict__ r2b,  // 1
                               float* __restrict__ out) {      // B
  int row = blockIdx.x * 256 + threadIdx.x;
  if (row >= NTOT) return;
  float acc = r2b[0];
  const bf16x8* tp = (const bf16x8*)(t2 + (long long)row * Hc);
#pragma unroll
  for (int c = 0; c < 16; ++c) {
    bf16x8 x = tp[c];
#pragma unroll
    for (int e = 0; e < 8; ++e) acc += (float)x[e] * r2w[c * 8 + e];
  }
  atomicAdd(&out[row / Nc], acc);
}

// ---------------- host ----------------
static inline size_t al256(size_t x) { return (x + 255) & ~(size_t)255; }

static void launch_gemm(const __bf16* A, const __bf16* Bp, const float* bias,
                        float* outF, __bf16* outB, int M, int N, int K, int siluFlag,
                        hipStream_t st) {
  int Mtiles = M / 16;
  dim3 grid((Mtiles + 7) / 8, N / 128);
  gemm_wmma_kernel<<<grid, 256, 0, st>>>(A, Bp, bias, outF, outB, N, K, siluFlag, Mtiles);
}

extern "C" void kernel_launch(void* const* d_in, const int* in_sizes, int n_in,
                              void* d_out, int out_size, void* d_ws, size_t ws_size,
                              hipStream_t stream) {
  const int*   nodes  = (const int*)d_in[0];
  const float* xyz    = (const float*)d_in[1];
  const int*   edges  = (const int*)d_in[2];
  const float* edisp  = (const float*)d_in[3];
  const float* cell   = (const float*)d_in[4];
  const float* emb    = (const float*)d_in[5];
  const float* filt_w = (const float*)d_in[6];
  const float* filt_b = (const float*)d_in[7];
  const float* m1_w   = (const float*)d_in[8];
  const float* m1_b   = (const float*)d_in[9];
  const float* m2_w   = (const float*)d_in[10];
  const float* m2_b   = (const float*)d_in[11];
  const float* u_w    = (const float*)d_in[12];
  const float* v_w    = (const float*)d_in[13];
  const float* c1_w   = (const float*)d_in[14];
  const float* c1_b   = (const float*)d_in[15];
  const float* c2_w   = (const float*)d_in[16];
  const float* c2_b   = (const float*)d_in[17];
  const float* r1_w   = (const float*)d_in[18];
  const float* r1_b   = (const float*)d_in[19];
  const float* r2_w   = (const float*)d_in[20];
  const float* r2_b   = (const float*)d_in[21];
  float* out = (float*)d_out;

  // ---- workspace carve-out ----
  char* w = (char*)d_ws;
  size_t off = 0;
  auto alloc = [&](size_t bytes) -> char* {
    char* p = w + off;
    off = al256(off + bytes);
    return p;
  };
  float*  unit  = (float*)alloc((size_t)ETOT * 3 * 4);
  float*  fc    = (float*)alloc((size_t)ETOT * 4);
  __bf16* esB   = (__bf16*)alloc((size_t)ETOT * 32 * 2);
  float*  s     = (float*)alloc((size_t)NTOT * Hc * 4);
  float*  v     = (float*)alloc((size_t)NTOT * 3 * Hc * 4);
  float*  v_old = (float*)alloc((size_t)NTOT * 3 * Hc * 4);
  __bf16* vB    = (__bf16*)alloc((size_t)NTOT * 3 * Hc * 2);
  __bf16* sgB   = (__bf16*)alloc((size_t)ETOT * Hc * 2);
  __bf16* tB    = (__bf16*)alloc((size_t)ETOT * Hc * 2);
  float*  U     = (float*)alloc((size_t)NTOT * 3 * Hc * 4);
  float*  V     = (float*)alloc((size_t)NTOT * 3 * Hc * 4);
  float*  UdotV = (float*)alloc((size_t)NTOT * Hc * 4);
  __bf16* catB  = (__bf16*)alloc((size_t)NTOT * 256 * 2);
  __bf16* a1B   = (__bf16*)alloc((size_t)NTOT * Hc * 2);
  float*  aBuf  = (float*)alloc((size_t)NTOT * 384 * 4);
  __bf16* sB    = (__bf16*)alloc((size_t)NTOT * Hc * 2);
  __bf16* t2B   = (__bf16*)alloc((size_t)NTOT * Hc * 2);
  __bf16* fwP   = (__bf16*)alloc((size_t)Lc * 32 * 384 * 2);
  __bf16* m1P   = (__bf16*)alloc((size_t)Lc * 128 * 128 * 2);
  __bf16* m2P   = (__bf16*)alloc((size_t)Lc * 128 * 384 * 2);
  __bf16* uwP   = (__bf16*)alloc((size_t)Lc * 128 * 128 * 2);
  __bf16* vwP   = (__bf16*)alloc((size_t)Lc * 128 * 128 * 2);
  __bf16* c1P   = (__bf16*)alloc((size_t)Lc * 256 * 128 * 2);
  __bf16* c2P   = (__bf16*)alloc((size_t)Lc * 128 * 384 * 2);
  __bf16* r1P   = (__bf16*)alloc((size_t)128 * 128 * 2);
  (void)ws_size; (void)in_sizes; (void)n_in; (void)out_size;

  auto packW = [&](const float* src, __bf16* dst, int Kp, int N, int Ksrc) {
    int total = Kp * N;
    pack_b_kernel<<<(total + 255) / 256, 256, 0, stream>>>(src, dst, Kp, N, Ksrc);
  };
  auto castN = [&](const float* src, __bf16* dst, long long n) {
    long long n8 = n / 8;
    cast_f32_bf16_kernel<<<(int)((n8 + 255) / 256), 256, 0, stream>>>(src, dst, n8);
  };

  // ---- one-time weight fragment packing ----
  for (int l = 0; l < Lc; ++l) {
    packW(filt_w + (size_t)l * EMBc * 384, fwP + (size_t)l * 32 * 384, 32, 384, EMBc);
    packW(m1_w + (size_t)l * 128 * 128, m1P + (size_t)l * 128 * 128, 128, 128, 128);
    packW(m2_w + (size_t)l * 128 * 384, m2P + (size_t)l * 128 * 384, 128, 384, 128);
    packW(u_w + (size_t)l * 128 * 128, uwP + (size_t)l * 128 * 128, 128, 128, 128);
    packW(v_w + (size_t)l * 128 * 128, vwP + (size_t)l * 128 * 128, 128, 128, 128);
    packW(c1_w + (size_t)l * 256 * 128, c1P + (size_t)l * 256 * 128, 256, 128, 256);
    packW(c2_w + (size_t)l * 128 * 384, c2P + (size_t)l * 128 * 384, 128, 384, 128);
  }
  packW(r1_w, r1P, 128, 128, 128);

  // ---- geometry + init ----
  edge_geom_kernel<<<(ETOT + 255) / 256, 256, 0, stream>>>(xyz, edges, edisp, cell,
                                                           unit, fc, esB);
  embed_kernel<<<(NTOT * 32 + 255) / 256, 256, 0, stream>>>(nodes, emb, s);
  zero_f32_kernel<<<(NTOT * 3 * 32 + 255) / 256, 256, 0, stream>>>(
      v, (long long)NTOT * 3 * 32);

  // ---- layers ----
  for (int l = 0; l < Lc; ++l) {
    long long ng = (long long)ETOT * 16;
    gather_cast_kernel<<<(int)((ng + 255) / 256), 256, 0, stream>>>(s, edges, sgB);
    // t = silu(sg @ m1_w + m1_b)  (bf16 out)
    launch_gemm(sgB, m1P + (size_t)l * 128 * 128, m1_b + l * 128, nullptr, tB,
                ETOT, 128, 128, 1, stream);
    // snapshot v for message construction
    hipMemcpyAsync(v_old, v, (size_t)NTOT * 3 * Hc * 4, hipMemcpyDeviceToDevice, stream);
    // fused: m2 GEMM + filter GEMM + fc + scatter
    edge_message_kernel<<<ETOT / 16, 256, 0, stream>>>(
        tB, esB, m2P + (size_t)l * 128 * 384, m2_b + l * 384,
        fwP + (size_t)l * 32 * 384, filt_b + l * 384, fc, unit, v_old, edges, s, v);
    // node stage
    castN(v, vB, (long long)NTOT * 3 * Hc);
    launch_gemm(vB, uwP + (size_t)l * 128 * 128, nullptr, U, nullptr,
                NTOT * 3, 128, 128, 0, stream);
    launch_gemm(vB, vwP + (size_t)l * 128 * 128, nullptr, V, nullptr,
                NTOT * 3, 128, 128, 0, stream);
    node_pre_kernel<<<(NTOT * 32 + 255) / 256, 256, 0, stream>>>(s, U, V, UdotV, catB);
    launch_gemm(catB, c1P + (size_t)l * 256 * 128, c1_b + l * 128, nullptr, a1B,
                NTOT, 128, 256, 1, stream);
    launch_gemm(a1B, c2P + (size_t)l * 128 * 384, c2_b + l * 384, aBuf, nullptr,
                NTOT, 384, 128, 0, stream);
    node_update_kernel<<<(NTOT * 32 + 255) / 256, 256, 0, stream>>>(aBuf, U, UdotV, s, v);
  }

  // ---- readout ----
  castN(s, sB, (long long)NTOT * Hc);
  launch_gemm(sB, r1P, r1_b, nullptr, t2B, NTOT, 128, 128, 1, stream);
  zero_f32_kernel<<<1, 256, 0, stream>>>(out, 2);
  readout_kernel<<<(NTOT + 255) / 256, 256, 0, stream>>>(t2B, r2_w, r2_b, out);
}
